// JK_43662637531875
// MI455X (gfx1250) — matrix operand order repaired and checked by reading, not compiled, not run on hardware
//
#include <hip/hip_runtime.h>
#include <hip/hip_bf16.h>

typedef __attribute__((ext_vector_type(16))) __bf16 v16bf;
typedef __attribute__((ext_vector_type(8)))  __bf16 v8bf;
typedef __attribute__((ext_vector_type(8)))  float  v8f;

__device__ __forceinline__ void atomAddF(float* p, float v) {
    __hip_atomic_fetch_add(p, v, __ATOMIC_RELAXED, __HIP_MEMORY_SCOPE_AGENT);
}

// ---------------- utility kernels ----------------
__global__ void fill_f32_kernel(float* __restrict__ p, float v, long n) {
    long t = (long)blockIdx.x * blockDim.x + threadIdx.x;
    if (t < n) p[t] = v;
}

__global__ void cvt_bf16_kernel(const float* __restrict__ in, __bf16* __restrict__ out, long n) {
    long t = (long)blockIdx.x * blockDim.x + threadIdx.x;
    if (t < n) out[t] = (__bf16)in[t];
}

__global__ void deg_accum_kernel(const int* __restrict__ dst, float* __restrict__ deg,
                                 int E, int N) {
    long e = (long)blockIdx.x * blockDim.x + threadIdx.x;
    if (e >= E) return;
    int d = dst[e];
    d = (d < 0) ? 0 : (d >= N ? N - 1 : d);
    atomAddF(&deg[d], 1.0f);
}

__global__ void rsqrt_kernel(float* __restrict__ p, int n) {
    long t = (long)blockIdx.x * blockDim.x + threadIdx.x;
    if (t < n) p[t] = rsqrtf(p[t]);
}

// ---------------- bf16 WMMA GEMM: C[M x Nlog] = A[M x K] * B[K x Nlog] (+bias) ----------------
// Block: 128 threads = 4 waves. Block tile = 64 rows x (NT*16) cols.
// Wave w owns rows [w*16, w*16+16) and sweeps all NT column tiles, so its A fragment
// (2 x global_load_b128 per K-step) is reused for NT WMMAs. NT is COMPILE-TIME so the
// accumulators acc[NT] stay in statically named VGPRs (no v_movrel indexing).
// B is transposed-staged into LDS once per block (contiguous K-runs -> ds_load_b128).
template <int NT>
__global__ __launch_bounds__(128) void gemm_bf16_kernel(
    const __bf16* __restrict__ A, const __bf16* __restrict__ B,
    const float* __restrict__ bias, float* __restrict__ C,
    int M, int K, int Nlog)
{
    constexpr int Npad = NT * 16;
    __shared__ __align__(16) __bf16 Bt[64 * 128];   // [Npad][K] transposed, max 16 KB

    const int tid = threadIdx.x;
    const int totalB = Npad * K;
    for (int idx = tid; idx < totalB; idx += 128) {
        int n = idx / K, k = idx - n * K;
        __bf16 v = (__bf16)0.0f;
        if (n < Nlog) v = B[(long)k * Nlog + n];
        Bt[n * K + k] = v;
    }
    __syncthreads();

    const int wave    = tid >> 5;
    const int lane    = tid & 31;
    const int hi      = lane >> 4;                  // 0: lanes 0-15, 1: lanes 16-31
    const int colLane = lane & 15;
    const int rowBase = blockIdx.x * 64 + wave * 16;

    int rA = rowBase + colLane;                     // (lane & 15) is the A row within the stripe
    if (rA >= M) rA = M - 1;
    const long aoff = (long)rA * K;

    v8f acc[NT] = {};
    for (int kb = 0; kb < K; kb += 32) {
        // A tile 16x32 bf16 striping: lanes0-15 elems0..7 = K kb+0..7, lanes16-31 = kb+8..15;
        // elems 8..15 are +16 further along K.
        const v8bf a_lo = *(const v8bf*)(A + aoff + kb + hi * 8);
        const v8bf a_hi = *(const v8bf*)(A + aoff + kb + 16 + hi * 8);
        v16bf a;
#pragma unroll
        for (int j = 0; j < 8; ++j) { a[j] = a_lo[j]; a[j + 8] = a_hi[j]; }

        if (kb + 32 < K)
            __builtin_prefetch(A + aoff + kb + 32, 0, 0);   // global_prefetch_b8

#pragma unroll
        for (int ct = 0; ct < NT; ++ct) {
            // B tile 32x16: lanes0-15 elems0..7 = K kb+0..7, lanes16-31 = kb+8..15; elems 8..15 are +16 further along K
            const __bf16* bp = &Bt[(ct * 16 + colLane) * K + kb + hi * 8];
            const v8bf b_lo = *(const v8bf*)(bp);
            const v8bf b_hi = *(const v8bf*)(bp + 16);
            v16bf b;
#pragma unroll
            for (int j = 0; j < 8; ++j) { b[j] = b_lo[j]; b[j + 8] = b_hi[j]; }

            acc[ct] = __builtin_amdgcn_wmma_f32_16x16x32_bf16(
                /*neg_a=*/false, a, /*neg_b=*/false, b,
                /*c_mod=*/(short)0, acc[ct], /*reuse_a=*/false, /*reuse_b=*/false);
        }
    }

#pragma unroll
    for (int ct = 0; ct < NT; ++ct) {
        const int col  = ct * 16 + colLane;
        const float badd = (bias != nullptr && col < Nlog) ? bias[col] : 0.0f;
#pragma unroll
        for (int v = 0; v < 8; ++v) {
            int r = rowBase + v + hi * 8;           // C/D layout: lanes16-31 hold M=8..15
            if (r < M && col < Nlog)
                C[(long)r * Nlog + col] = acc[ct][v] + badd;
        }
    }
}

// ---------------- edge scatter: agg[dst] += h[src] * dinv[src]*dinv[dst] ----------------
// 16 lanes per edge, each handles a float4 chunk of the 64-wide row.
__global__ __launch_bounds__(256) void scatter_kernel(
    const float* __restrict__ h, const int* __restrict__ src, const int* __restrict__ dst,
    const float* __restrict__ dinv, float* __restrict__ agg, int E, int N)
{
    long t = (long)blockIdx.x * blockDim.x + threadIdx.x;
    long e = t >> 4;
    int  j = (int)(t & 15);
    if (e >= E) return;
    int s = src[e]; s = (s < 0) ? 0 : (s >= N ? N - 1 : s);
    int d = dst[e]; d = (d < 0) ? 0 : (d >= N ? N - 1 : d);
    float nm = dinv[s] * dinv[d];
    const float4 hv = *(const float4*)(h + (long)s * 64 + j * 4);
    float* ap = agg + (long)d * 64 + j * 4;
    atomAddF(ap + 0, hv.x * nm);
    atomAddF(ap + 1, hv.y * nm);
    atomAddF(ap + 2, hv.z * nm);
    atomAddF(ap + 3, hv.w * nm);
}

// ---------------- finish layer 1: x1 = relu(agg + h*dinv^2 + b); emit f32 + bf16 ----------------
__global__ void finish1_kernel(const float* __restrict__ agg, const float* __restrict__ h,
                               const float* __restrict__ dinv, const float* __restrict__ bias,
                               float* __restrict__ x1, __bf16* __restrict__ x1b, long total)
{
    long t = (long)blockIdx.x * blockDim.x + threadIdx.x;
    if (t >= total) return;
    int i = (int)(t >> 6), j = (int)(t & 63);
    float di = dinv[i];
    float r = fmaxf(agg[t] + h[t] * di * di + bias[j], 0.0f);
    x1[t] = r;
    x1b[t] = (__bf16)r;
}

// ---------------- finish layer 2 fused with JK-max: xjk = max(x1, relu(...)) -> bf16 ----------------
__global__ void finish2_kernel(const float* __restrict__ agg, const float* __restrict__ h,
                               const float* __restrict__ dinv, const float* __restrict__ bias,
                               const float* __restrict__ x1, __bf16* __restrict__ xjkb, long total)
{
    long t = (long)blockIdx.x * blockDim.x + threadIdx.x;
    if (t >= total) return;
    int i = (int)(t >> 6), j = (int)(t & 63);
    float di = dinv[i];
    float r2 = fmaxf(agg[t] + h[t] * di * di + bias[j], 0.0f);
    xjkb[t] = (__bf16)fmaxf(x1[t], r2);
}

// ---------------- host orchestration ----------------
extern "C" void kernel_launch(void* const* d_in, const int* in_sizes, int n_in,
                              void* d_out, int out_size, void* d_ws, size_t ws_size,
                              hipStream_t stream) {
    const float* x   = (const float*)d_in[0];
    const int*   ei  = (const int*)d_in[1];
    const float* W1  = (const float*)d_in[2];
    const float* b1  = (const float*)d_in[3];
    const float* W2  = (const float*)d_in[4];
    const float* b2  = (const float*)d_in[5];
    const float* Wf  = (const float*)d_in[6];
    const float* bfv = (const float*)d_in[7];

    const int NF = 128, NH = 64, NC = 40;
    const int N  = in_sizes[0] / NF;
    const int E  = in_sizes[1] / 2;
    const int* src = ei;
    const int* dst = ei + E;

    // carve workspace (256B aligned)
    char* ws = (char*)d_ws;
    size_t off = 0;
    auto carve = [&](size_t bytes) -> char* {
        char* p = ws + off;
        off += (bytes + 255) & ~(size_t)255;
        return p;
    };
    float*  dinv = (float*) carve((size_t)N * 4);
    __bf16* xb   = (__bf16*)carve((size_t)N * NF * 2);  // layer-1 input; reused for x1b / xjkb
    float*  h    = (float*) carve((size_t)N * NH * 4);
    float*  agg  = (float*) carve((size_t)N * NH * 4);
    float*  x1   = (float*) carve((size_t)N * NH * 4);
    __bf16* w1b  = (__bf16*)carve((size_t)NF * NH * 2);
    __bf16* w2b  = (__bf16*)carve((size_t)NH * NH * 2);
    __bf16* wfb  = (__bf16*)carve((size_t)NH * NC * 2);
    (void)ws_size; (void)n_in; (void)out_size;

    const long nX  = (long)N * NF;
    const long nH  = (long)N * NH;
    const int  TB  = 256;
    auto blks = [](long n, int b) { return (int)((n + b - 1) / b); };

    // convert inputs/weights to bf16
    cvt_bf16_kernel<<<blks(nX, TB), TB, 0, stream>>>(x, xb, nX);
    cvt_bf16_kernel<<<blks((long)NF * NH, TB), TB, 0, stream>>>(W1, w1b, (long)NF * NH);
    cvt_bf16_kernel<<<blks((long)NH * NH, TB), TB, 0, stream>>>(W2, w2b, (long)NH * NH);
    cvt_bf16_kernel<<<blks((long)NH * NC, TB), TB, 0, stream>>>(Wf, wfb, (long)NH * NC);

    // degrees (self-loop folded in via init=1.0) -> dinv
    fill_f32_kernel<<<blks(N, TB), TB, 0, stream>>>(dinv, 1.0f, N);
    deg_accum_kernel<<<blks(E, TB), TB, 0, stream>>>(dst, dinv, E, N);
    rsqrt_kernel<<<blks(N, TB), TB, 0, stream>>>(dinv, N);

    const int gemmGrid = (N + 63) / 64;   // 64 rows per block

    // ---- layer 1 ----
    gemm_bf16_kernel<4><<<gemmGrid, 128, 0, stream>>>(xb, w1b, nullptr, h, N, NF, NH);
    fill_f32_kernel<<<blks(nH, TB), TB, 0, stream>>>(agg, 0.0f, nH);
    scatter_kernel<<<blks((long)E * 16, TB), TB, 0, stream>>>(h, src, dst, dinv, agg, E, N);
    finish1_kernel<<<blks(nH, TB), TB, 0, stream>>>(agg, h, dinv, b1, x1, xb, nH);

    // ---- layer 2 ----
    gemm_bf16_kernel<4><<<gemmGrid, 128, 0, stream>>>(xb, w2b, nullptr, h, N, NH, NH);
    fill_f32_kernel<<<blks(nH, TB), TB, 0, stream>>>(agg, 0.0f, nH);
    scatter_kernel<<<blks((long)E * 16, TB), TB, 0, stream>>>(h, src, dst, dinv, agg, E, N);
    finish2_kernel<<<blks(nH, TB), TB, 0, stream>>>(agg, h, dinv, b2, x1, xb, nH);

    // ---- final linear: out = xjk @ Wf + bf  (N=40, 3 column tiles = 48 padded) ----
    gemm_bf16_kernel<3><<<gemmGrid, 128, 0, stream>>>(xb, wfb, bfv, (float*)d_out, N, NH, NC);
}
